// Vit_44641890075150
// MI455X (gfx1250) — compile-verified, matrix operand order checked by reading, not run on hardware
//
#include <hip/hip_runtime.h>
#include <hip/hip_bf16.h>

// ---------------------------------------------------------------------------
// ViT forward for gfx1250 (MI455X). bf16 WMMA GEMMs, f32 accumulate.
// GEMM: 128x128 block tile, 8 waves each computing 32x64 (2x4 WMMA tiles),
// double-buffered LDS fed by GLOBAL_LOAD_ASYNC_TO_LDS_B128 (ASYNCcnt),
// one barrier per K-step, fused bias/GELU/bf16 output.
// ---------------------------------------------------------------------------

typedef unsigned short u16;
typedef unsigned int   u32;
typedef __attribute__((ext_vector_type(16))) __bf16 v16bf;
typedef __attribute__((ext_vector_type(8)))  float  v8f;

struct FragPair { uint4 lo, hi; };
union FragU { FragPair p; v16bf v; };

#define LDS_PITCH 40   // 32 bf16 + 8 pad -> 80B row pitch, conflict-free b128 reads

// ----------------------------- constants -----------------------------------
#define VB    32          // batch
#define VE    768         // embed
#define VL    6           // layers
#define VH    8           // heads
#define VHD   96          // head dim
#define VN    197         // tokens
#define VNP   196         // patches
#define VFF   3072        // mlp hidden
#define VNC   1000        // classes
#define VTOK  (VB * VN)   // 6304 token rows
#define VZ    (VB * VH)   // 256 attention batches
#define VKP   224         // padded attention K (197 -> 224, mult of 32)
#define VSCALE 0.102062072615966f   // 96^-0.5, applied AFTER softmax (faithful)

__device__ __forceinline__ u16 f2bf(float f) {
    u32 u = __float_as_uint(f);
    u32 r = u + 0x7FFFu + ((u >> 16) & 1u);   // round-to-nearest-even
    return (u16)(r >> 16);
}
__device__ __forceinline__ float gelu_exact(float x) {
    return 0.5f * x * (1.0f + erff(x * 0.70710678118654752f));
}

// CDNA5 async global->LDS copy (16B per lane), ASYNCcnt-tracked.
// lds = byte offset within workgroup LDS (generic ptr truncated: ISA 10.2),
// g   = per-lane global address.
__device__ __forceinline__ void async_ld_b128(u32 lds, const void* g) {
    asm volatile("global_load_async_to_lds_b128 %0, %1, off"
                 :: "v"(lds), "v"((unsigned long long)(uintptr_t)g)
                 : "memory");
}
__device__ __forceinline__ void async_wait0() {
    asm volatile("s_wait_asynccnt 0x0" ::: "memory");
}
__device__ __forceinline__ u32 lds_off(const void* p) {
    return (u32)(uintptr_t)p;   // LDS aperture: addr[31:0] is the LDS address
}

// ----------------------------- elementwise ----------------------------------
__global__ void k_cvt_bf16(const float* __restrict__ a, u16* __restrict__ o, long n) {
    long i = (long)blockIdx.x * blockDim.x + threadIdx.x;
    if (i < n) o[i] = f2bf(a[i]);
}

// im2col: patches[b*196+p, c*256+i*16+j] = x[b,c,16*hh+i,16*ww+j]
__global__ void k_im2col(const float* __restrict__ x, u16* __restrict__ patches) {
    long idx = (long)blockIdx.x * blockDim.x + threadIdx.x;
    long total = (long)VB * VNP * VE;
    if (idx >= total) return;
    int col = (int)(idx % VE);
    long row = idx / VE;
    int b = (int)(row / VNP), p = (int)(row % VNP);
    int hh = p / 14, ww = p % 14;
    int c = col >> 8, rem = col & 255, i = rem >> 4, j = rem & 15;
    float v = x[(((long)b * 3 + c) * 224 + hh * 16 + i) * 224 + ww * 16 + j];
    patches[idx] = f2bf(v);
}

// x[b,0,:]=cls ; x[b,1+p,:]=embed[b*196+p,:] ; += pos ; write f32 + bf16
__global__ void k_assemble(const float* __restrict__ embed,
                           const float* __restrict__ cls,
                           const float* __restrict__ pos,
                           float* __restrict__ xf, u16* __restrict__ xb) {
    long idx = (long)blockIdx.x * blockDim.x + threadIdx.x;
    long total = (long)VB * VN * VE;
    if (idx >= total) return;
    int e = (int)(idx % VE);
    long rn = idx / VE;
    int n = (int)(rn % VN);
    int b = (int)(rn / VN);
    float v = (n == 0) ? cls[e] : embed[((long)b * VNP + (n - 1)) * VE + e];
    v += pos[(long)n * VE + e];
    xf[idx] = v;
    xb[idx] = f2bf(v);
}

// split qkv -> q,k [z, n, 96] bf16 ; vT [z, 96, 224] bf16 (cols 197..223 = 0)
__global__ void k_split_qkv(const float* __restrict__ qkv,
                            u16* __restrict__ q, u16* __restrict__ k,
                            u16* __restrict__ vT) {
    long idx = (long)blockIdx.x * blockDim.x + threadIdx.x;
    long total = (long)VZ * VKP * VHD;
    if (idx >= total) return;
    int d  = (int)(idx % VHD);
    int nn = (int)((idx / VHD) % VKP);
    int z  = (int)(idx / ((long)VHD * VKP));
    int b = z >> 3, h = z & 7;
    float vv = 0.f;
    if (nn < VN) {
        long base = ((long)(b * VN + nn)) * (3 * VE) + h * VHD + d;
        q[((long)z * VN + nn) * VHD + d] = f2bf(qkv[base]);
        k[((long)z * VN + nn) * VHD + d] = f2bf(qkv[base + VE]);
        vv = qkv[base + 2 * VE];
    }
    vT[((long)z * VHD + d) * VKP + nn] = f2bf(vv);
}

// softmax over 197 cols, * SCALE after, bf16 out padded to 224 (pad = 0)
__global__ void k_softmax(const float* __restrict__ sim, u16* __restrict__ probs) {
    __shared__ float red[256];
    int t = threadIdx.x;
    long row = blockIdx.x;                     // z*197+n
    const float* s = sim + row * VN;
    u16* p = probs + row * VKP;
    float v = (t < VN) ? s[t] : -3.4e38f;
    red[t] = v; __syncthreads();
    for (int st = 128; st > 0; st >>= 1) { if (t < st) red[t] = fmaxf(red[t], red[t + st]); __syncthreads(); }
    float mx = red[0]; __syncthreads();
    float e = (t < VN) ? __expf(v - mx) : 0.f;
    red[t] = e; __syncthreads();
    for (int st = 128; st > 0; st >>= 1) { if (t < st) red[t] += red[t + st]; __syncthreads(); }
    float inv = VSCALE / red[0];
    if (t < VKP) p[t] = f2bf((t < VN) ? e * inv : 0.f);
}

// x = LN(x + d) * g + b  (E=768, 3 elems/thread) ; writes f32 + bf16
__global__ void k_add_ln(float* __restrict__ xf, const float* __restrict__ d,
                         const float* __restrict__ g, const float* __restrict__ bb,
                         u16* __restrict__ xb) {
    __shared__ float r1[256], r2[256];
    int t = threadIdx.x;
    long row = blockIdx.x;
    float* xr = xf + row * VE;
    const float* dr = d + row * VE;
    float v0 = xr[t] + dr[t];
    float v1 = xr[t + 256] + dr[t + 256];
    float v2 = xr[t + 512] + dr[t + 512];
    r1[t] = v0 + v1 + v2;
    r2[t] = v0 * v0 + v1 * v1 + v2 * v2;
    __syncthreads();
    for (int st = 128; st > 0; st >>= 1) {
        if (t < st) { r1[t] += r1[t + st]; r2[t] += r2[t + st]; }
        __syncthreads();
    }
    float mean = r1[0] * (1.0f / VE);
    float var  = r2[0] * (1.0f / VE) - mean * mean;
    float rst  = rsqrtf(var + 1e-5f);
    #pragma unroll
    for (int q = 0; q < 3; ++q) {
        int e = t + q * 256;
        float v = (q == 0 ? v0 : (q == 1 ? v1 : v2));
        float o = (v - mean) * rst * g[e] + bb[e];
        xr[e] = o;
        xb[row * VE + e] = f2bf(o);
    }
}

// mean pool over N=197 then LN(head) -> bf16 [32, 768]
__global__ void k_pool_ln(const float* __restrict__ xf,
                          const float* __restrict__ g, const float* __restrict__ bb,
                          u16* __restrict__ poolb) {
    __shared__ float r1[256], r2[256];
    int t = threadIdx.x, b = blockIdx.x;
    float a0 = 0.f, a1 = 0.f, a2 = 0.f;
    for (int n = 0; n < VN; ++n) {
        const float* r = xf + ((long)(b * VN + n)) * VE;
        a0 += r[t]; a1 += r[t + 256]; a2 += r[t + 512];
    }
    const float inv = 1.0f / VN;
    a0 *= inv; a1 *= inv; a2 *= inv;
    r1[t] = a0 + a1 + a2;
    r2[t] = a0 * a0 + a1 * a1 + a2 * a2;
    __syncthreads();
    for (int st = 128; st > 0; st >>= 1) {
        if (t < st) { r1[t] += r1[t + st]; r2[t] += r2[t + st]; }
        __syncthreads();
    }
    float mean = r1[0] * (1.0f / VE);
    float var  = r2[0] * (1.0f / VE) - mean * mean;
    float rst  = rsqrtf(var + 1e-5f);
    #pragma unroll
    for (int q = 0; q < 3; ++q) {
        int e = t + q * 256;
        float a = (q == 0 ? a0 : (q == 1 ? a1 : a2));
        poolb[(long)b * VE + e] = f2bf((a - mean) * rst * g[e] + bb[e]);
    }
}

// ----------------------------- WMMA GEMM ------------------------------------
// C[M,Nw] = A[M,K](bf16) * W[Nw,K](bf16)^T + bias ; optional exact GELU.
// Outputs: C (f32, skipped if null) and/or Cb (bf16, skipped if null).
// Batched: z = blockIdx.z; A += z*sA ; W += z*sB ;
//          C += (z/cdiv)*sCo + (z%cdiv)*sCi    (cdiv=1 -> plain batch).
// Block = 256 thr = 8 waves as 4(M)x2(N); wave tile 32x64 = 2x4 WMMA tiles.
// Block tile 128x128, K step 32, double-buffered LDS filled by async copies.
__global__ void __launch_bounds__(256)
k_gemm_bf16(const u16* __restrict__ A, const u16* __restrict__ W,
            const float* __restrict__ bias, float* __restrict__ C,
            u16* __restrict__ Cb,
            int M, int Nw, int K, int ldc,
            long sA, long sB, int cdiv, long sCo, long sCi, int act) {
    __shared__ u16 Asub[2][128 * LDS_PITCH];
    __shared__ u16 Bsub[2][128 * LDS_PITCH];

    const int tid  = threadIdx.x;
    const int lane = tid & 31;
    const int wv   = tid >> 5;
    const int wm   = wv & 3;           // wave M sub-tile (32 rows each)
    const int wn   = wv >> 2;          // wave N sub-tile (64 cols each)
    const int m0   = blockIdx.x * 128;
    const int n0   = blockIdx.y * 128;
    const int z    = blockIdx.z;

    const u16* Ab = A + (long)z * sA;
    const u16* Wb = W + (long)z * sB;

    // cooperative staging: A 128x32 and B 128x32, 16 bf16 (32B) per thread each
    const int srow  = tid >> 1;             // 0..127
    const int shalf = (tid & 1) << 4;       // 0 or 16 (bf16 elems)
    int arg = m0 + srow; if (arg >= M) arg = M - 1;
    int brg = n0 + srow; if (brg >= Nw) brg = Nw - 1;

    const int rsel = (lane & 16) ? 8 : 0;
    const int la   = lane & 15;

    auto load_tiles = [&](int kk, int pb) {
        const u16* ag = Ab + (long)arg * K + kk + shalf;
        u32 al = lds_off(&Asub[pb][srow * LDS_PITCH + shalf]);
        async_ld_b128(al,      ag);
        async_ld_b128(al + 16, ag + 8);
        const u16* bg = Wb + (long)brg * K + kk + shalf;
        u32 bl = lds_off(&Bsub[pb][srow * LDS_PITCH + shalf]);
        async_ld_b128(bl,      bg);
        async_ld_b128(bl + 16, bg + 8);
        if (kk + 32 < K) {
            __builtin_prefetch(ag + 32, 0, 0);
            __builtin_prefetch(bg + 32, 0, 0);
        }
    };

    v8f cc[2][4];
    #pragma unroll
    for (int i = 0; i < 2; ++i)
        #pragma unroll
        for (int j = 0; j < 4; ++j) cc[i][j] = v8f{};

    int p = 0;
    load_tiles(0, 0);
    async_wait0();
    __syncthreads();

    for (int kk = 0; kk < K; kk += 32) {
        int np = p ^ 1;
        if (kk + 32 < K) load_tiles(kk + 32, np);   // async, overlaps WMMAs

        const u16* ap0 = &Asub[p][(wm * 32 + la) * LDS_PITCH + rsel];
        const u16* ap1 = ap0 + 16 * LDS_PITCH;
        FragU a0, a1;
        a0.p.lo = *reinterpret_cast<const uint4*>(ap0);
        a0.p.hi = *reinterpret_cast<const uint4*>(ap0 + 16);
        a1.p.lo = *reinterpret_cast<const uint4*>(ap1);
        a1.p.hi = *reinterpret_cast<const uint4*>(ap1 + 16);

        const u16* bbase = &Bsub[p][(wn * 64 + la) * LDS_PITCH + rsel];
        #pragma unroll
        for (int j = 0; j < 4; ++j) {
            const u16* bp = bbase + j * (16 * LDS_PITCH);
            FragU b;
            b.p.lo = *reinterpret_cast<const uint4*>(bp);
            b.p.hi = *reinterpret_cast<const uint4*>(bp + 16);
            cc[0][j] = __builtin_amdgcn_wmma_f32_16x16x32_bf16(
                false, a0.v, false, b.v, (short)0, cc[0][j], false, false);
            cc[1][j] = __builtin_amdgcn_wmma_f32_16x16x32_bf16(
                false, a1.v, false, b.v, (short)0, cc[1][j], false, false);
        }

        async_wait0();        // next tile landed in LDS
        __syncthreads();
        p = np;
    }

    // store: VGPR r -> M = r (lanes 0-15) / 8+r (lanes 16-31), N = lane&15
    long cz = (long)(z / cdiv) * sCo + (long)(z % cdiv) * sCi;
    const int mb = m0 + wm * 32 + ((lane & 16) ? 8 : 0);
    const int nbase = n0 + wn * 64 + la;
    #pragma unroll
    for (int j = 0; j < 4; ++j) {
        int n_c = nbase + j * 16;
        if (n_c >= Nw) continue;
        float bv = bias ? bias[n_c] : 0.f;
        #pragma unroll
        for (int i = 0; i < 2; ++i) {
            int mrow = mb + i * 16;
            #pragma unroll
            for (int r = 0; r < 8; ++r) {
                int m = mrow + r;
                if (m < M) {
                    float v = cc[i][j][r] + bv;
                    if (act) v = gelu_exact(v);
                    long off = cz + (long)m * ldc + n_c;
                    if (C)  C[off]  = v;
                    if (Cb) Cb[off] = f2bf(v);
                }
            }
        }
    }
}

// ----------------------------- workspace map --------------------------------
static constexpr size_t AL(size_t x) { return (x + 255) & ~(size_t)255; }
static constexpr size_t OFF_WQKV  = 0;                                               // 6*2304*768 bf16
static constexpr size_t OFF_WPROJ = AL(OFF_WQKV  + (size_t)10616832 * 2);
static constexpr size_t OFF_WM1   = AL(OFF_WPROJ + (size_t)3538944  * 2);
static constexpr size_t OFF_WM2   = AL(OFF_WM1   + (size_t)14155776 * 2);
static constexpr size_t OFF_WCONV = AL(OFF_WM2   + (size_t)14155776 * 2);
static constexpr size_t OFF_WHEAD = AL(OFF_WCONV + (size_t)589824   * 2);
static constexpr size_t OFF_PATCH = AL(OFF_WHEAD + (size_t)768000   * 2);            // 6272x768 bf16
static constexpr size_t OFF_XF    = AL(OFF_PATCH + (size_t)4816896  * 2);            // 6304x768 f32
static constexpr size_t OFF_XB    = AL(OFF_XF    + (size_t)4841472  * 4);            // bf16
static constexpr size_t OFF_QKV   = AL(OFF_XB    + (size_t)4841472  * 2);            // 6304x2304 f32
static constexpr size_t OFF_Q     = AL(OFF_QKV   + (size_t)14524416 * 4);            // 256x197x96 bf16
static constexpr size_t OFF_K     = AL(OFF_Q     + (size_t)4841472  * 2);
static constexpr size_t OFF_VT    = AL(OFF_K     + (size_t)4841472  * 2);            // 256x96x224 bf16
static constexpr size_t OFF_SIM   = AL(OFF_VT    + (size_t)5505024  * 2);            // 256x197x197 f32
static constexpr size_t OFF_PROB  = AL(OFF_SIM   + (size_t)9935104  * 4);            // 256x197x224 bf16
static constexpr size_t OFF_T0    = AL(OFF_PROB  + (size_t)11296768 * 2);            // 6272x768 f32 (embed)
static constexpr size_t OFF_T0B   = AL(OFF_T0    + (size_t)4841472  * 4);            // 6304x768 bf16 (attn out)
static constexpr size_t OFF_T1    = AL(OFF_T0B   + (size_t)4841472  * 2);            // 6304x768 f32
static constexpr size_t OFF_HDNB  = AL(OFF_T1    + (size_t)4841472  * 4);            // 6304x3072 bf16
static constexpr size_t OFF_POOLB = AL(OFF_HDNB  + (size_t)19365888 * 2);            // 32x768 bf16

// ----------------------------- launch ----------------------------------------
static inline void cvt(const float* s, u16* d, long n, hipStream_t st) {
    k_cvt_bf16<<<(int)((n + 255) / 256), 256, 0, st>>>(s, d, n);
}

extern "C" void kernel_launch(void* const* d_in, const int* in_sizes, int n_in,
                              void* d_out, int out_size, void* d_ws, size_t ws_size,
                              hipStream_t stream) {
    const float* x        = (const float*)d_in[0];
    const float* conv_w   = (const float*)d_in[1];
    const float* conv_b   = (const float*)d_in[2];
    const float* cls_tok  = (const float*)d_in[3];
    const float* pos_emb  = (const float*)d_in[4];
    const float* qkv_w    = (const float*)d_in[5];
    const float* qkv_b    = (const float*)d_in[6];
    const float* proj_w   = (const float*)d_in[7];
    const float* proj_b   = (const float*)d_in[8];
    const float* ln1_g    = (const float*)d_in[9];
    const float* ln1_b    = (const float*)d_in[10];
    const float* mlp_w1   = (const float*)d_in[11];
    const float* mlp_b1   = (const float*)d_in[12];
    const float* mlp_w2   = (const float*)d_in[13];
    const float* mlp_b2   = (const float*)d_in[14];
    const float* ln2_g    = (const float*)d_in[15];
    const float* ln2_b    = (const float*)d_in[16];
    const float* hln_g    = (const float*)d_in[17];
    const float* hln_b    = (const float*)d_in[18];
    const float* head_w   = (const float*)d_in[19];
    const float* head_b   = (const float*)d_in[20];
    float* out = (float*)d_out;

    char* ws = (char*)d_ws;
    u16*   wqkv  = (u16*)(ws + OFF_WQKV);
    u16*   wproj = (u16*)(ws + OFF_WPROJ);
    u16*   wm1   = (u16*)(ws + OFF_WM1);
    u16*   wm2   = (u16*)(ws + OFF_WM2);
    u16*   wconv = (u16*)(ws + OFF_WCONV);
    u16*   whead = (u16*)(ws + OFF_WHEAD);
    u16*   patch = (u16*)(ws + OFF_PATCH);
    float* xf    = (float*)(ws + OFF_XF);
    u16*   xb    = (u16*)(ws + OFF_XB);
    float* qkv   = (float*)(ws + OFF_QKV);
    u16*   qb    = (u16*)(ws + OFF_Q);
    u16*   kb    = (u16*)(ws + OFF_K);
    u16*   vT    = (u16*)(ws + OFF_VT);
    float* sim   = (float*)(ws + OFF_SIM);
    u16*   prob  = (u16*)(ws + OFF_PROB);
    float* t0    = (float*)(ws + OFF_T0);
    u16*   t0b   = (u16*)(ws + OFF_T0B);
    float* t1    = (float*)(ws + OFF_T1);
    u16*   hdnb  = (u16*)(ws + OFF_HDNB);
    u16*   poolb = (u16*)(ws + OFF_POOLB);

    // ---- weights to bf16 (every launch: deterministic) ----
    cvt(qkv_w,  wqkv,  10616832, stream);
    cvt(proj_w, wproj, 3538944,  stream);
    cvt(mlp_w1, wm1,   14155776, stream);
    cvt(mlp_w2, wm2,   14155776, stream);
    cvt(conv_w, wconv, 589824,   stream);
    cvt(head_w, whead, 768000,   stream);

    // ---- patch embed ----
    {
        long n = (long)VB * VNP * VE;
        k_im2col<<<(int)((n + 255) / 256), 256, 0, stream>>>(x, patch);
    }
    {   // embed[6272,768] = patch * conv_w^T + conv_b   -> t0 (f32)
        dim3 g((VB * VNP + 127) / 128, (VE + 127) / 128, 1);
        k_gemm_bf16<<<g, 256, 0, stream>>>(patch, wconv, conv_b, t0, nullptr,
                                           VB * VNP, VE, VE, VE,
                                           0, 0, 1, 0, 0, 0);
    }
    {
        long n = (long)VTOK * VE;
        k_assemble<<<(int)((n + 255) / 256), 256, 0, stream>>>(t0, cls_tok, pos_emb, xf, xb);
    }

    // ---- transformer layers ----
    for (int l = 0; l < VL; ++l) {
        const u16* wq = wqkv  + (size_t)l * 3 * VE * VE;
        const u16* wp = wproj + (size_t)l * VE * VE;
        const u16* w1 = wm1   + (size_t)l * VFF * VE;
        const u16* w2 = wm2   + (size_t)l * VFF * VE;

        // qkv = x @ qw^T + qb : [6304, 2304] (f32, consumed by split)
        {
            dim3 g((VTOK + 127) / 128, (3 * VE) / 128, 1);
            k_gemm_bf16<<<g, 256, 0, stream>>>(xb, wq, qkv_b + (size_t)l * 3 * VE, qkv, nullptr,
                                               VTOK, 3 * VE, VE, 3 * VE,
                                               0, 0, 1, 0, 0, 0);
        }
        // split -> q,k [z,197,96], vT [z,96,224] (zero-padded)
        {
            long n = (long)VZ * VKP * VHD;
            k_split_qkv<<<(int)((n + 255) / 256), 256, 0, stream>>>(qkv, qb, kb, vT);
        }
        // sim[z,197,197] = q @ k^T (batched over 256)
        {
            dim3 g((VN + 127) / 128, (VN + 127) / 128, VZ);
            k_gemm_bf16<<<g, 256, 0, stream>>>(qb, kb, nullptr, sim, nullptr,
                                               VN, VN, VHD, VN,
                                               (long)VN * VHD, (long)VN * VHD,
                                               1, (long)VN * VN, 0, 0);
        }
        // softmax (*SCALE after) -> probs bf16 [z,197,224]
        k_softmax<<<VZ * VN, 256, 0, stream>>>(sim, prob);
        // att merged [b,n,h*96+d] = probs @ vT^T  -> t0b (bf16 direct)
        {
            dim3 g((VN + 127) / 128, (VHD + 127) / 128, VZ);
            k_gemm_bf16<<<g, 256, 0, stream>>>(prob, vT, nullptr, nullptr, t0b,
                                               VN, VHD, VKP, VE,
                                               (long)VN * VKP, (long)VHD * VKP,
                                               VH, (long)VN * VE, (long)VHD, 0);
        }
        // proj : t1 = att @ pw^T + pb (f32 for residual)
        {
            dim3 g((VTOK + 127) / 128, VE / 128, 1);
            k_gemm_bf16<<<g, 256, 0, stream>>>(t0b, wp, proj_b + (size_t)l * VE, t1, nullptr,
                                               VTOK, VE, VE, VE,
                                               0, 0, 1, 0, 0, 0);
        }
        // x = LN(x + t1)
        k_add_ln<<<VTOK, 256, 0, stream>>>(xf, t1, ln1_g + (size_t)l * VE,
                                           ln1_b + (size_t)l * VE, xb);
        // mlp hidden = gelu(x @ w1^T + b1) : bf16 direct [6304, 3072]
        {
            dim3 g((VTOK + 127) / 128, VFF / 128, 1);
            k_gemm_bf16<<<g, 256, 0, stream>>>(xb, w1, mlp_b1 + (size_t)l * VFF, nullptr, hdnb,
                                               VTOK, VFF, VE, VFF,
                                               0, 0, 1, 0, 0, 1);
        }
        // mlp out = hdn @ w2^T + b2 : [6304, 768] f32
        {
            dim3 g((VTOK + 127) / 128, VE / 128, 1);
            k_gemm_bf16<<<g, 256, 0, stream>>>(hdnb, w2, mlp_b2 + (size_t)l * VE, t1, nullptr,
                                               VTOK, VE, VFF, VE,
                                               0, 0, 1, 0, 0, 0);
        }
        // x = LN(x + mlp)
        k_add_ln<<<VTOK, 256, 0, stream>>>(xf, t1, ln2_g + (size_t)l * VE,
                                           ln2_b + (size_t)l * VE, xb);
    }

    // ---- head: mean pool -> LN -> linear ----
    k_pool_ln<<<VB, 256, 0, stream>>>(xf, hln_g, hln_b, poolb);
    {
        dim3 g((VB + 127) / 128, (VNC + 127) / 128, 1);
        k_gemm_bf16<<<g, 256, 0, stream>>>(poolb, whead, head_b, out, nullptr,
                                           VB, VNC, VE, VNC,
                                           0, 0, 1, 0, 0, 0);
    }
    (void)in_sizes; (void)n_in; (void)out_size; (void)ws_size;
}
